// ConformerBlock_67053029425286
// MI455X (gfx1250) — matrix operand order repaired and checked by reading, not compiled, not run on hardware
//
#include <hip/hip_runtime.h>
#include <hip/hip_bf16.h>
#include <math.h>

typedef __attribute__((ext_vector_type(16))) __bf16 v16bf;
typedef __attribute__((ext_vector_type(8)))  float  v8f;
typedef __attribute__((ext_vector_type(4)))  unsigned int v4u;
typedef __attribute__((ext_vector_type(8)))  int v8i32;
typedef __attribute__((ext_vector_type(4)))  int v4i32;

namespace {

constexpr int Dm    = 512;
constexpr int Tn    = 1024;
constexpr int Hh    = 8;
constexpr int HDd   = 64;
constexpr int BT    = 8192;   // B*T
constexpr int DF    = 2048;   // E*D
constexpr int INNER = 1024;   // CE*D
constexpr int MAXP  = 512;
constexpr int NP    = 1025;   // 2*MAXP+1
constexpr int KW    = 31;
constexpr float EPS = 1e-5f;

__device__ __forceinline__ float sigm(float x){ return 1.f/(1.f+__expf(-x)); }

// ----------------- conversion / packing kernels -----------------
__global__ void k_cvt_bf16(const float* __restrict__ s, __bf16* __restrict__ d, int n){
  int i = blockIdx.x*256 + threadIdx.x;
  if (i < n) d[i] = (__bf16)s[i];
}
// transpose-convert: src f32 [K][N] -> dst bf16 [N][K]
__global__ void k_cvt_t(const float* __restrict__ s, __bf16* __restrict__ d, int K, int N){
  int i = blockIdx.x*256 + threadIdx.x;
  if (i >= K*N) return;
  int n = i / K, kk = i % K;
  d[i] = (__bf16)s[(size_t)kk*N + n];
}
// merged qkv weight, transposed: dst [1536][512]
__global__ void k_qkv_w(const float* __restrict__ wq, const float* __restrict__ wkv,
                        __bf16* __restrict__ w){
  int i = blockIdx.x*256 + threadIdx.x;
  if (i >= Dm*1536) return;
  int n = i / Dm, k = i % Dm;
  float v = (n < Dm) ? wq[(size_t)k*Dm + n] : wkv[(size_t)k*(2*Dm) + (n-Dm)];
  w[i] = (__bf16)v;
}
__global__ void k_qkv_b(const float* __restrict__ bq, const float* __restrict__ bkv,
                        float* __restrict__ bb){
  int i = blockIdx.x*256 + threadIdx.x;
  if (i < 1536) bb[i] = (i < Dm) ? bq[i] : bkv[i-Dm];
}

// ----------------- layernorm (one wave32 per row of 512) -----------------
template<bool OUTF>
__global__ __launch_bounds__(128) void k_ln(const float* __restrict__ x,
    const float* __restrict__ g, const float* __restrict__ b,
    __bf16* __restrict__ yb, float* __restrict__ yf){
  int wave = threadIdx.x >> 5, lane = threadIdx.x & 31;
  int row  = blockIdx.x*4 + wave;
  const float* xr = x + (size_t)row*Dm;
  float v[16]; float s = 0.f;
  #pragma unroll
  for (int i=0;i<16;++i){ v[i] = xr[lane + i*32]; s += v[i]; }
  #pragma unroll
  for (int o=16;o;o>>=1) s += __shfl_xor(s, o, 32);
  float mean = s * (1.f/Dm);
  float q = 0.f;
  #pragma unroll
  for (int i=0;i<16;++i){ float d0 = v[i]-mean; q += d0*d0; }
  #pragma unroll
  for (int o=16;o;o>>=1) q += __shfl_xor(q, o, 32);
  float inv = rsqrtf(q*(1.f/Dm) + EPS);
  #pragma unroll
  for (int i=0;i<16;++i){
    int c = lane + i*32;
    float y = (v[i]-mean)*inv*g[c] + b[c];
    if (OUTF) yf[(size_t)row*Dm + c] = y;
    else      yb[(size_t)row*Dm + c] = (__bf16)y;
  }
}

// ----------------- WMMA fragment helpers (16x16x32 bf16) -----------------
// A (16xK, 16-bit): lanes 0-15 row=lane, elems 0-15 -> K 0-15;
// lanes 16-31 row=lane-16, elems 0-15 -> K 16-31.
__device__ __forceinline__ v16bf frag_a_lds(const __bf16* base, int ld, int row0, int kofs){
  int lane = threadIdx.x & 31;
  int r  = row0 + (lane & 15);
  int kb = (lane < 16) ? 0 : 16;
  v16bf f;
  #pragma unroll
  for (int e=0;e<8;++e) f[e]   = base[r*ld + kofs + kb + e];      // 16B contiguous
  #pragma unroll
  for (int e=0;e<8;++e) f[8+e] = base[r*ld + kofs + 8 + kb + e]; // 16B contiguous
  return f;
}
// B (32x16): source tile stored K-transposed in LDS as [n][32 k] -> contiguous 32B/lane.
__device__ __forceinline__ v16bf frag_b_ldsT(const __bf16* base, int col0){
  int lane = threadIdx.x & 31;
  int n  = col0 + (lane & 15);
  int kb = (lane < 16) ? 0 : 16;
  v16bf f;
  #pragma unroll
  for (int e=0;e<16;++e) f[e] = base[n*32 + kb + e];              // 32B contiguous
  return f;
}

// ----------------- Tensor Data Mover: 2D bf16 tile (nrows x 32 elems) -> LDS ----------
// D# group0: count=1 | lds_addr | 57b global_addr | type=2 ("image")
// D# group1: data_size=2B, tensor_dim0/1, tile_dim0=32, tile_dim1=nrows, dim0_stride
__device__ __forceinline__ void tdm_load_tile(const __bf16* gsrc, const __bf16* lds_dst,
                                              unsigned dim0, unsigned dim1, unsigned stride,
                                              unsigned nrows){
  unsigned long long ga = (unsigned long long)(uintptr_t)gsrc;
  unsigned lds_off = (unsigned)(uintptr_t)lds_dst;  // generic LDS ptr: low 32b = LDS offset
  v4u g0;
  g0[0] = 1u;                                        // count=1 valid descriptor
  g0[1] = lds_off;
  g0[2] = (unsigned)(ga & 0xFFFFFFFFu);
  g0[3] = (unsigned)((ga >> 32) & 0x01FFFFFFu) | (2u << 30);
  v8i32 g1;
  g1[0] = (int)(1u << 16);                           // data_size = 2 bytes
  g1[1] = (int)((dim0 & 0xFFFFu) << 16);             // tensor_dim0[15:0]
  g1[2] = (int)((dim0 >> 16) | ((dim1 & 0xFFFFu) << 16));
  g1[3] = (int)((dim1 >> 16) | (32u << 16));         // tile_dim0 = 32 elems
  g1[4] = (int)(nrows & 0xFFFFu);                    // tile_dim1
  g1[5] = (int)stride;                               // tensor_dim0_stride[31:0]
  g1[6] = 0;
  g1[7] = 0;
  v4i32 z4 = {0,0,0,0};
  v8i32 z8 = {0,0,0,0,0,0,0,0};
  // 6-arg form on this toolchain: (g0, g1, g2, g3, ext, cpol)
  __builtin_amdgcn_tensor_load_to_lds(g0, g1, z4, z4, z8, 0);
}

enum { ACT_NONE = 0, ACT_SILU = 1 };

// ----------------- generic bf16 WMMA GEMM: out = epi(A[M,K] @ W[K,N] + bias) ---------
// Weight passed TRANSPOSED: Wt[N][K]. 128 threads (4 waves), 64x64 tile.
// TDM stages both 64x32 operand tiles into LDS; waves run 16x16x32 WMMAs.
template<int ACT, bool RES, bool OUTBF>
__global__ __launch_bounds__(128) void k_gemm(
    const __bf16* __restrict__ A, const __bf16* __restrict__ Wt,
    const float* __restrict__ bias, const float* __restrict__ res, float alpha,
    float* __restrict__ outF, __bf16* __restrict__ outB, int M, int N, int K){
  __shared__ __bf16 sA[64*32];   // A tile, row-major [64 m][32 k]
  __shared__ __bf16 sW[64*32];   // W tile, [64 n][32 k]
  const int tid = threadIdx.x, wave = tid>>5, lane = tid&31;
  const int m0 = blockIdx.y*64, n0 = blockIdx.x*64;
  v8f acc[4] = {};
  for (int k0=0; k0<K; k0+=32){
    if (wave == 0){
      tdm_load_tile(A  + (size_t)m0*K + k0, sA, (unsigned)K, (unsigned)M, (unsigned)K, 64);
      tdm_load_tile(Wt + (size_t)n0*K + k0, sW, (unsigned)K, (unsigned)N, (unsigned)K, 64);
      __builtin_amdgcn_s_wait_tensorcnt(0);
    }
    if (k0 + 32 < K){
      __builtin_prefetch(A  + (size_t)(m0 + (tid>>1))*K + k0 + 32, 0, 1);
      __builtin_prefetch(Wt + (size_t)(n0 + (tid>>1))*K + k0 + 32, 0, 1);
    }
    __syncthreads();
    v16bf af = frag_a_lds(sA, 32, wave*16, 0);
    #pragma unroll
    for (int j=0;j<4;++j){
      v16bf bfr = frag_b_ldsT(sW, j*16);
      acc[j] = __builtin_amdgcn_wmma_f32_16x16x32_bf16(false, af, false, bfr,
                                                       (short)0, acc[j], false, false);
    }
    __syncthreads();
  }
  // epilogue: C/D layout -> lanes 0-15: rows v, lanes 16-31: rows 8+v; col = lane&15
  const int nl = lane & 15, mb = (lane<16)?0:8;
  #pragma unroll
  for (int j=0;j<4;++j){
    int col = n0 + j*16 + nl;
    float bv = bias[col];
    #pragma unroll
    for (int v=0;v<8;++v){
      int row = m0 + wave*16 + mb + v;
      float val = acc[j][v] + bv;
      if (ACT == ACT_SILU) val = val * sigm(val);
      if (RES) val = res[(size_t)row*N + col] + alpha*val;
      if (OUTBF) outB[(size_t)row*N + col] = (__bf16)val;
      else       outF[(size_t)row*N + col] = val;
    }
  }
}

// ----------------- flash attention w/ relative position bias ----------------
// one wave per (b, h, 16-row tile); qkv layout [BT,1536]: q|k|v sections of 512.
__global__ __launch_bounds__(32) void k_attn(const __bf16* __restrict__ qkv,
                                             const __bf16* __restrict__ pe,
                                             __bf16* __restrict__ outO){
  __shared__ __bf16 sQ[16*64];
  __shared__ __bf16 sRB[16*1026];   // rb[i][p] = q_i . pe_p  (bf16, ~33KB)
  __shared__ float  sS[16*32];
  __shared__ float  sP[16*32];
  __shared__ float  rowM[16], rowL[16], rowC[16];
  const int lane = threadIdx.x & 31;
  const int tpb  = Tn/16;
  const int t16  = blockIdx.x % tpb;
  const int h    = (blockIdx.x / tpb) % Hh;
  const int b    = blockIdx.x / (tpb*Hh);
  const int n0   = t16*16;
  const size_t rowstr = 3*Dm;  // 1536
  { // load Q tile [16,64]
    int r = lane >> 1, half = lane & 1;
    const __bf16* src = qkv + (size_t)(b*Tn + n0 + r)*rowstr + h*HDd + half*32;
    *(v16bf*)&sQ[r*64 + half*32]      = *(const v16bf*)src;
    *(v16bf*)&sQ[r*64 + half*32 + 16] = *(const v16bf*)(src + 16);
  }
  if (lane < 16){ rowM[lane] = -1e30f; rowL[lane] = 0.f; }
  __syncthreads();
  v16bf aq0 = frag_a_lds(sQ, 64, 0, 0);
  v16bf aq1 = frag_a_lds(sQ, 64, 0, 32);
  const int nl = lane & 15, mb = (lane<16)?0:8, kb16 = (lane<16)?0:16;
  // rb = Q @ pe^T  (WMMA over 65 column tiles)
  for (int p0=0; p0<=1024; p0+=16){
    int p = p0 + nl; if (p > 1024) p = 1024;
    v8f c = {};
    #pragma unroll
    for (int ch=0; ch<2; ++ch){
      v16bf bfr;
      #pragma unroll
      for (int e=0;e<16;++e) bfr[e] = pe[(size_t)p*HDd + ch*32 + kb16 + e];
      c = __builtin_amdgcn_wmma_f32_16x16x32_bf16(false, ch?aq1:aq0, false, bfr,
                                                  (short)0, c, false, false);
    }
    if (p0 + nl <= 1024){
      #pragma unroll
      for (int v=0;v<8;++v) sRB[(mb+v)*1026 + p0 + nl] = (__bf16)c[v];
    }
  }
  __syncthreads();
  v8f o[4] = {};
  for (int m0=0; m0<Tn; m0+=32){
    #pragma unroll
    for (int jt=0;jt<2;++jt){               // S = Q K^T (two 16-col tiles)
      v8f s = {};
      int m = m0 + jt*16 + nl;
      #pragma unroll
      for (int ch=0; ch<2; ++ch){
        const __bf16* kr = qkv + (size_t)(b*Tn + m)*rowstr + Dm + h*HDd + ch*32 + kb16;
        v16bf bfr;
        #pragma unroll
        for (int e=0;e<16;++e) bfr[e] = kr[e];
        s = __builtin_amdgcn_wmma_f32_16x16x32_bf16(false, ch?aq1:aq0, false, bfr,
                                                    (short)0, s, false, false);
      }
      #pragma unroll
      for (int v=0;v<8;++v) sS[(mb+v)*32 + jt*16 + nl] = s[v];
    }
    __syncthreads();
    if (lane < 16){                         // online softmax, rel-bias gather
      int r = lane, gn = n0 + r;
      float mx = rowM[r];
      float vals[32];
      #pragma unroll
      for (int c=0;c<32;++c){
        int d0 = gn - (m0 + c);
        if (d0 >  MAXP) d0 =  MAXP;
        if (d0 < -MAXP) d0 = -MAXP;
        float val = (sS[r*32+c] + (float)sRB[r*1026 + d0 + MAXP]) * 0.125f;
        vals[c] = val; mx = fmaxf(mx, val);
      }
      float corr = __expf(rowM[r] - mx);
      float l = rowL[r] * corr;
      #pragma unroll
      for (int c=0;c<32;++c){ float pv = __expf(vals[c]-mx); sP[r*32+c] = pv; l += pv; }
      rowM[r] = mx; rowL[r] = l; rowC[r] = corr;
    }
    __syncthreads();
    float cf[8];
    #pragma unroll
    for (int v=0;v<8;++v) cf[v] = rowC[mb+v];
    #pragma unroll
    for (int j=0;j<4;++j)
      #pragma unroll
      for (int v=0;v<8;++v) o[j][v] *= cf[v];
    v16bf pf;                               // P tile -> A-layout (bf16)
    {
      int r = lane & 15, kb8 = (lane<16)?0:16;
      #pragma unroll
      for (int e=0;e<8;++e) pf[e]   = (__bf16)sP[r*32 + kb8 + e];
      #pragma unroll
      for (int e=0;e<8;++e) pf[8+e] = (__bf16)sP[r*32 + 8 + kb8 + e];
    }
    #pragma unroll
    for (int j=0;j<4;++j){                  // O += P @ V
      v16bf vf;
      int dcol = 2*Dm + h*HDd + j*16 + nl;
      #pragma unroll
      for (int e=0;e<16;++e)
        vf[e] = qkv[(size_t)(b*Tn + m0 + kb16 + e)*rowstr + dcol];
      o[j] = __builtin_amdgcn_wmma_f32_16x16x32_bf16(false, pf, false, vf,
                                                     (short)0, o[j], false, false);
    }
    __syncthreads();
  }
  float il[8];
  #pragma unroll
  for (int v=0;v<8;++v) il[v] = 1.f / rowL[mb+v];
  #pragma unroll
  for (int j=0;j<4;++j)
    #pragma unroll
    for (int v=0;v<8;++v){
      int row = b*Tn + n0 + mb + v;
      int col = h*HDd + j*16 + nl;
      outO[(size_t)row*Dm + col] = (__bf16)(o[j][v] * il[v]);
    }
}

// ----------------- conv-module elementwise / reduction kernels -----------------
__global__ void k_glu(const float* __restrict__ hc, float* __restrict__ g){
  int i = blockIdx.x*256 + threadIdx.x;
  if (i >= BT*INNER) return;
  int r = i / INNER, c = i % INNER;
  float a  = hc[(size_t)r*DF + c];
  float gt = hc[(size_t)r*DF + INNER + c];
  g[i] = a * sigm(gt);
}
__global__ void k_dwconv(const float* __restrict__ g, const float* __restrict__ dw,
                         const float* __restrict__ db, float* __restrict__ out){
  int i = blockIdx.x*256 + threadIdx.x;
  if (i >= BT*INNER) return;
  int c = i % INNER, bt = i / INNER;
  int b = bt / Tn, t = bt % Tn;
  float acc = db[c];
  #pragma unroll
  for (int k=0;k<KW;++k){
    int tt = t + k - (KW-1)/2;
    if (tt >= 0 && tt < Tn) acc += g[(size_t)(b*Tn + tt)*INNER + c] * dw[c*KW + k];
  }
  out[i] = acc;
}
__global__ __launch_bounds__(256) void k_bnstats(const float* __restrict__ x,
                                                 float* __restrict__ stats){
  __shared__ float ss[256], sq[256];
  int c = blockIdx.x;
  float s = 0.f, q = 0.f;
  for (int r = threadIdx.x; r < BT; r += 256){
    float v = x[(size_t)r*INNER + c];
    s += v; q += v*v;
  }
  ss[threadIdx.x] = s; sq[threadIdx.x] = q;
  __syncthreads();
  for (int o=128;o;o>>=1){
    if (threadIdx.x < o){ ss[threadIdx.x]+=ss[threadIdx.x+o]; sq[threadIdx.x]+=sq[threadIdx.x+o]; }
    __syncthreads();
  }
  if (threadIdx.x==0){ stats[c]=ss[0]; stats[INNER+c]=sq[0]; }
}
__global__ void k_bnsilu(const float* __restrict__ x, const float* __restrict__ stats,
                         const float* __restrict__ gg, const float* __restrict__ bb,
                         __bf16* __restrict__ y){
  int i = blockIdx.x*256 + threadIdx.x;
  if (i >= BT*INNER) return;
  int c = i % INNER;
  float m   = stats[c] * (1.f/BT);
  float var = stats[INNER+c] * (1.f/BT) - m*m;
  float inv = rsqrtf(var + EPS);
  float v = (x[i]-m)*inv*gg[c] + bb[c];
  y[i] = (__bf16)(v * sigm(v));
}

} // anonymous namespace

extern "C" void kernel_launch(void* const* d_in, const int* in_sizes, int n_in,
                              void* d_out, int out_size, void* d_ws, size_t ws_size,
                              hipStream_t stream){
  (void)in_sizes; (void)n_in; (void)out_size; (void)ws_size;
  const float* x    = (const float*)d_in[0];
  const float* f1g  = (const float*)d_in[1];  const float* f1b  = (const float*)d_in[2];
  const float* f1w1 = (const float*)d_in[3];  const float* f1b1 = (const float*)d_in[4];
  const float* f1w2 = (const float*)d_in[5];  const float* f1b2 = (const float*)d_in[6];
  const float* png  = (const float*)d_in[7];  const float* pnb  = (const float*)d_in[8];
  const float* wq   = (const float*)d_in[9];  const float* bq   = (const float*)d_in[10];
  const float* wkv  = (const float*)d_in[11]; const float* bkv  = (const float*)d_in[12];
  const float* wo   = (const float*)d_in[13]; const float* bo   = (const float*)d_in[14];
  const float* pe   = (const float*)d_in[15];
  const float* clg  = (const float*)d_in[16]; const float* clb  = (const float*)d_in[17];
  const float* cw1  = (const float*)d_in[18]; const float* cb1  = (const float*)d_in[19];
  const float* cdw  = (const float*)d_in[20]; const float* cdb  = (const float*)d_in[21];
  const float* cbg  = (const float*)d_in[22]; const float* cbb  = (const float*)d_in[23];
  const float* cw2  = (const float*)d_in[24]; const float* cb2  = (const float*)d_in[25];
  const float* f2g  = (const float*)d_in[26]; const float* f2b  = (const float*)d_in[27];
  const float* f2w1 = (const float*)d_in[28]; const float* f2b1 = (const float*)d_in[29];
  const float* f2w2 = (const float*)d_in[30]; const float* f2b2 = (const float*)d_in[31];
  const float* pog  = (const float*)d_in[32]; const float* pob  = (const float*)d_in[33];
  float* outp = (float*)d_out;

  // ---- workspace carve-up (~190MB; regions reused across pipeline stages) ----
  char* ws = (char*)d_ws;
  size_t off = 0;
  auto alloc = [&](size_t bytes)->char*{
    char* p = ws + off; off += (bytes + 255) & ~(size_t)255; return p;
  };
  __bf16* W1   = (__bf16*)alloc((size_t)Dm*DF*2);      // all weights stored [N][K] bf16
  __bf16* W2   = (__bf16*)alloc((size_t)DF*Dm*2);
  __bf16* WQKV = (__bf16*)alloc((size_t)Dm*1536*2);
  __bf16* WO   = (__bf16*)alloc((size_t)Dm*Dm*2);
  __bf16* WC1  = (__bf16*)alloc((size_t)Dm*DF*2);
  __bf16* WC2  = (__bf16*)alloc((size_t)INNER*Dm*2);
  __bf16* W3   = (__bf16*)alloc((size_t)Dm*DF*2);
  __bf16* W4   = (__bf16*)alloc((size_t)DF*Dm*2);
  __bf16* PEb  = (__bf16*)alloc((size_t)NP*HDd*2);
  float*  BQKV = (float*) alloc(1536*4);
  float*  STAT = (float*) alloc(2*INNER*4);
  float*  XA   = (float*) alloc((size_t)BT*Dm*4);
  float*  XB   = (float*) alloc((size_t)BT*Dm*4);
  __bf16* LNB  = (__bf16*)alloc((size_t)BT*Dm*2);
  __bf16* HBF  = (__bf16*)alloc((size_t)BT*DF*2);      // H1/H2 bf16; later CN bf16
  char*   UN1  = alloc((size_t)BT*INNER*4);            // QKV bf16 (24MB) then G f32 (32MB)
  __bf16* AO   = (__bf16*)alloc((size_t)BT*Dm*2);
  float*  UN2  = (float*) alloc((size_t)BT*DF*4);      // HC f32 (64MB) then conv-out f32
  __bf16* QKV  = (__bf16*)UN1;
  float*  Gf   = (float*)UN1;
  float*  HC   = UN2;
  float*  Cf   = UN2;
  __bf16* CN   = HBF;

  const int EW   = (BT*INNER + 255)/256;   // elementwise grid over 8M elems
  const dim3 blk128(128), blk256(256), blk32(32);

  // ---- weights -> bf16, transposed to [N][K] ----
  k_cvt_t<<<(Dm*DF+255)/256,    blk256, 0, stream>>>(f1w1, W1,  Dm,    DF);
  k_cvt_t<<<(DF*Dm+255)/256,    blk256, 0, stream>>>(f1w2, W2,  DF,    Dm);
  k_cvt_t<<<(Dm*Dm+255)/256,    blk256, 0, stream>>>(wo,   WO,  Dm,    Dm);
  k_cvt_t<<<(Dm*DF+255)/256,    blk256, 0, stream>>>(cw1,  WC1, Dm,    DF);
  k_cvt_t<<<(INNER*Dm+255)/256, blk256, 0, stream>>>(cw2,  WC2, INNER, Dm);
  k_cvt_t<<<(Dm*DF+255)/256,    blk256, 0, stream>>>(f2w1, W3,  Dm,    DF);
  k_cvt_t<<<(DF*Dm+255)/256,    blk256, 0, stream>>>(f2w2, W4,  DF,    Dm);
  k_cvt_bf16<<<(NP*HDd+255)/256, blk256, 0, stream>>>(pe,  PEb, NP*HDd);
  k_qkv_w<<<(Dm*1536+255)/256,  blk256, 0, stream>>>(wq, wkv, WQKV);
  k_qkv_b<<<(1536+255)/256,     blk256, 0, stream>>>(bq, bkv, BQKV);

  // ---- stage 1: x1 = x + 0.5*FFN1(x) ----
  k_ln<false><<<BT/4, blk128, 0, stream>>>(x, f1g, f1b, LNB, nullptr);
  k_gemm<ACT_SILU,false,true><<<dim3(DF/64, BT/64), blk128, 0, stream>>>(
      LNB, W1, f1b1, nullptr, 0.f, nullptr, HBF, BT, DF, Dm);
  k_gemm<ACT_NONE,true,false><<<dim3(Dm/64, BT/64), blk128, 0, stream>>>(
      HBF, W2, f1b2, x, 0.5f, XA, nullptr, BT, Dm, DF);

  // ---- stage 2: x2 = x1 + Attn(LN(x1)) ----
  k_ln<false><<<BT/4, blk128, 0, stream>>>(XA, png, pnb, LNB, nullptr);
  k_gemm<ACT_NONE,false,true><<<dim3(1536/64, BT/64), blk128, 0, stream>>>(
      LNB, WQKV, BQKV, nullptr, 0.f, nullptr, QKV, BT, 1536, Dm);
  k_attn<<<8*Hh*(Tn/16), blk32, 0, stream>>>(QKV, PEb, AO);
  k_gemm<ACT_NONE,true,false><<<dim3(Dm/64, BT/64), blk128, 0, stream>>>(
      AO, WO, bo, XA, 1.0f, XB, nullptr, BT, Dm, Dm);

  // ---- stage 3: x3 = x2 + Conv(x2) ----
  k_ln<false><<<BT/4, blk128, 0, stream>>>(XB, clg, clb, LNB, nullptr);
  k_gemm<ACT_NONE,false,false><<<dim3(DF/64, BT/64), blk128, 0, stream>>>(
      LNB, WC1, cb1, nullptr, 0.f, HC, nullptr, BT, DF, Dm);
  k_glu<<<EW, blk256, 0, stream>>>(HC, Gf);
  k_dwconv<<<EW, blk256, 0, stream>>>(Gf, cdw, cdb, Cf);
  k_bnstats<<<INNER, blk256, 0, stream>>>(Cf, STAT);
  k_bnsilu<<<EW, blk256, 0, stream>>>(Cf, STAT, cbg, cbb, CN);
  k_gemm<ACT_NONE,true,false><<<dim3(Dm/64, BT/64), blk128, 0, stream>>>(
      CN, WC2, cb2, XB, 1.0f, XA, nullptr, BT, Dm, INNER);

  // ---- stage 4: x4 = x3 + 0.5*FFN2(x3) ----
  k_ln<false><<<BT/4, blk128, 0, stream>>>(XA, f2g, f2b, LNB, nullptr);
  k_gemm<ACT_SILU,false,true><<<dim3(DF/64, BT/64), blk128, 0, stream>>>(
      LNB, W3, f2b1, nullptr, 0.f, nullptr, HBF, BT, DF, Dm);
  k_gemm<ACT_NONE,true,false><<<dim3(Dm/64, BT/64), blk128, 0, stream>>>(
      HBF, W4, f2b2, XA, 0.5f, XB, nullptr, BT, Dm, DF);

  // ---- final LN -> d_out (f32) ----
  k_ln<true><<<BT/4, blk128, 0, stream>>>(XB, pog, pob, nullptr, outp);
}
